// CtrDNNModel_71957882077786
// MI455X (gfx1250) — compile-verified, matrix operand order checked by reading, not run on hardware
//
#include <hip/hip_runtime.h>
#include <cstdint>
#include <cstddef>

// ---------------------------------------------------------------------------
// CTR DNN on gfx1250: gather+pool (HBM-bound, ~36us roofline) followed by a
// bf16 WMMA MLP (v_wmma_f32_16x16x32_bf16, f32 accumulate).
// v2: 16x64 output tile per wave (4 accumulators, A-fragment reuse),
//     float4 gather loads, vectorized final layer, weight prefetch.
// ---------------------------------------------------------------------------

typedef __attribute__((ext_vector_type(16))) __bf16 v16bf;
typedef __attribute__((ext_vector_type(8)))  float  v8f;

#define EMB   128
#define BATCH 16384
#define BAG   50

// ---------------------------------------------------------------------------
// Kernel 1: embedding gather + mean pool.
// One WAVE per (batch, field) pair; each lane owns 4 consecutive emb dims and
// loads them as one b128 -> each row read is a single coalesced 512B burst
// (32 lanes x 16B). 8 waves / block. Output X0 bf16 [BATCH, 256].
// ---------------------------------------------------------------------------
__global__ __launch_bounds__(256) void pool_kernel(const int* __restrict__ ids,
                                                   const float* __restrict__ emb,
                                                   __bf16* __restrict__ X0) {
  __shared__ int sidx[8][BAG];
  const int lane = threadIdx.x & 31;
  const int wave = threadIdx.x >> 5;
  const int blk  = blockIdx.x * 8 + wave;        // (b,f) index, 0..BATCH*2-1
  const int b = blk >> 1;
  const int f = blk & 1;

  const int* base = ids + (size_t)blk * BAG;
  if (lane < 25) {                                // 50 indices in two strided halves
    sidx[wave][lane]      = base[lane];
    sidx[wave][lane + 25] = base[lane + 25];
  }
  // Producer and consumer are the same wave: LDS ordering via DScnt, no barrier.

  float ax = 0.0f, ay = 0.0f, az = 0.0f, aw = 0.0f;
  #pragma unroll 10
  for (int j = 0; j < BAG; ++j) {
    const float4 v =
        reinterpret_cast<const float4*>(emb + (size_t)sidx[wave][j] * EMB)[lane];
    ax += v.x; ay += v.y; az += v.z; aw += v.w;
  }
  const float s = 1.0f / (float)BAG;
  union { __bf16 h[4]; uint2 u; } o;
  o.h[0] = (__bf16)(ax * s);
  o.h[1] = (__bf16)(ay * s);
  o.h[2] = (__bf16)(az * s);
  o.h[3] = (__bf16)(aw * s);
  reinterpret_cast<uint2*>(X0 + (size_t)b * (2 * EMB) + f * EMB)[lane] = o.u;
}

// ---------------------------------------------------------------------------
// Kernel 2: fp32 -> bf16 weight conversion (weights arrive as fp32).
// ---------------------------------------------------------------------------
__global__ void f32_to_bf16(const float* __restrict__ src,
                            __bf16* __restrict__ dst, int n) {
  int i = blockIdx.x * blockDim.x + threadIdx.x;
  if (i < n) dst[i] = (__bf16)src[i];
}

// ---------------------------------------------------------------------------
// Kernel 3: GEMM + bias + ReLU via v_wmma_f32_16x16x32_bf16.
//   Y[M,N] = relu(X[M,K] @ W[N,K]^T + bias)     (bf16 in, bf16 out, f32 acc)
// One wave computes a 16(M) x 64(N) strip: 4 accumulators sharing one A
// fragment per K-step (4x A reuse, 12.8 flop/byte from cache).
//
// Fragment layouts per CDNA5 ISA 7.12.2 (16-bit A 16x32, wave32):
//   lanes 0..15  : row M = lane,    K = {0..7, 16..23}
//   lanes 16..31 : row M = lane-16, K = {8..15, 24..31}
// -> two contiguous 16B chunks per lane (global_load_b128 x2).
// B columns are W rows (x @ W.T) -> identical per-lane addressing.
// C/D: VGPR i -> row = i + 8*(lane>>4), col = lane&15.
// ---------------------------------------------------------------------------
union Frag {
  v16bf v;
  uint4 q[2];
};

template <int RELU>
__global__ __launch_bounds__(256) void gemm_bias_act(
    const __bf16* __restrict__ X,    // [M,K]
    const __bf16* __restrict__ W,    // [N,K]
    const float*  __restrict__ bias, // [N]
    __bf16* __restrict__ Y,          // [M,N]
    int M, int N, int K) {
  const int lane = threadIdx.x & 31;
  const int wave = threadIdx.x >> 5;            // 8 waves / block
  const int mbase = (blockIdx.x * 8 + wave) * 16;
  const int nbase = blockIdx.y * 64;

  const int r  = lane & 15;                     // A row / B col within tile
  const int kh = (lane >> 4) << 3;              // K sub-offset: 0 or 8

  const uint4* ap = reinterpret_cast<const uint4*>(X + (size_t)(mbase + r) * K + kh);
  const uint4* bp = reinterpret_cast<const uint4*>(W + (size_t)(nbase + r) * K + kh);
  const int bstride = 2 * K;                    // 16 rows * K elems / 8-per-uint4

  // Weight panel is small & hot: pull it toward the WGP ahead of the K loop.
  __builtin_prefetch(bp, 0, 3);
  __builtin_prefetch(bp + bstride, 0, 3);
  __builtin_prefetch(bp + 2 * bstride, 0, 3);
  __builtin_prefetch(bp + 3 * bstride, 0, 3);

  v8f c[4];
  #pragma unroll
  for (int t = 0; t < 4; ++t) {
    const float bv = bias[nbase + t * 16 + r];  // lane's 8 D elems share a col
    #pragma unroll
    for (int i = 0; i < 8; ++i) c[t][i] = bv;
  }

  for (int k = 0; k < K; k += 32) {
    Frag a;
    a.q[0] = ap[0];                             // K = kh .. kh+7
    a.q[1] = ap[2];                             // K = kh+16 .. kh+23
    ap += 4;                                    // advance 32 bf16
    #pragma unroll
    for (int t = 0; t < 4; ++t) {
      Frag bf;
      bf.q[0] = bp[t * bstride];
      bf.q[1] = bp[t * bstride + 2];
      c[t] = __builtin_amdgcn_wmma_f32_16x16x32_bf16(
          /*neg_a=*/false, a.v, /*neg_b=*/false, bf.v,
          /*c_mod=*/(short)0, c[t], /*reuse_a=*/false, /*reuse_b=*/false);
    }
    bp += 4;
  }

  const int rowbase = mbase + ((lane >> 4) << 3);
  #pragma unroll
  for (int t = 0; t < 4; ++t) {
    const int col = nbase + t * 16 + r;
    #pragma unroll
    for (int i = 0; i < 8; ++i) {
      float v = c[t][i];
      if (RELU) v = v > 0.0f ? v : 0.0f;
      Y[(size_t)(rowbase + i) * N + col] = (__bf16)v;
    }
  }
}

// ---------------------------------------------------------------------------
// Kernel 4: final 64 -> 1 dot product + sigmoid, fp32 out. Vectorized b128
// loads (8 bf16 per load); W5 reads are wave-uniform (scalar-cached).
// ---------------------------------------------------------------------------
__global__ void final_layer(const __bf16* __restrict__ X,   // [M,64]
                            const float* __restrict__ W5,   // [64]
                            const float* __restrict__ b5,   // [1]
                            float* __restrict__ out, int M) {
  int m = blockIdx.x * blockDim.x + threadIdx.x;
  if (m >= M) return;
  const uint4* xp = reinterpret_cast<const uint4*>(X + (size_t)m * 64);
  float acc = b5[0];
  #pragma unroll
  for (int cidx = 0; cidx < 8; ++cidx) {
    union { uint4 q; __bf16 h[8]; } u;
    u.q = xp[cidx];
    #pragma unroll
    for (int j = 0; j < 8; ++j) acc += (float)u.h[j] * W5[cidx * 8 + j];
  }
  out[m] = 1.0f / (1.0f + __expf(-acc));
}

// ---------------------------------------------------------------------------
// Launch.  Workspace layout (bytes), buffers recycled as layers retire:
//   X0: [0,   8M)   16384x256 bf16
//   X1: [8M, 24M)   16384x512 bf16
//   X2: [0,   8M)   16384x256 bf16  (X0 dead)
//   X3: [8M, 12M)   16384x128 bf16  (X1 dead)
//   X4: [12M,14M)   16384x64  bf16
//   Wb1..Wb4: [24M, ~24.6M)  converted bf16 weights
// ---------------------------------------------------------------------------
extern "C" void kernel_launch(void* const* d_in, const int* in_sizes, int n_in,
                              void* d_out, int out_size, void* d_ws, size_t ws_size,
                              hipStream_t stream) {
  const int*   ids = (const int*)  d_in[0];
  const float* emb = (const float*)d_in[1];
  const float* W1  = (const float*)d_in[2];  const float* b1 = (const float*)d_in[3];
  const float* W2  = (const float*)d_in[4];  const float* b2 = (const float*)d_in[5];
  const float* W3  = (const float*)d_in[6];  const float* b3 = (const float*)d_in[7];
  const float* W4  = (const float*)d_in[8];  const float* b4 = (const float*)d_in[9];
  const float* W5  = (const float*)d_in[10]; const float* b5 = (const float*)d_in[11];
  float* out = (float*)d_out;

  uint8_t* ws = (uint8_t*)d_ws;
  const size_t MB = 1u << 20;
  __bf16* X0  = (__bf16*)(ws + 0 * MB);
  __bf16* X1  = (__bf16*)(ws + 8 * MB);
  __bf16* X2  = (__bf16*)(ws + 0 * MB);
  __bf16* X3  = (__bf16*)(ws + 8 * MB);
  __bf16* X4  = (__bf16*)(ws + 12 * MB);
  __bf16* Wb1 = (__bf16*)(ws + 24 * MB);
  __bf16* Wb2 = (__bf16*)(ws + 24 * MB + 262144);
  __bf16* Wb3 = (__bf16*)(ws + 24 * MB + 2 * 262144);
  __bf16* Wb4 = (__bf16*)(ws + 24 * MB + 2 * 262144 + 65536);

  // Weight conversion (tiny; overlapped at stream head).
  f32_to_bf16<<<(512 * 256 + 255) / 256, 256, 0, stream>>>(W1, Wb1, 512 * 256);
  f32_to_bf16<<<(256 * 512 + 255) / 256, 256, 0, stream>>>(W2, Wb2, 256 * 512);
  f32_to_bf16<<<(128 * 256 + 255) / 256, 256, 0, stream>>>(W3, Wb3, 128 * 256);
  f32_to_bf16<<<( 64 * 128 + 255) / 256, 256, 0, stream>>>(W4, Wb4,  64 * 128);

  // Gather + mean pool (dominant cost: ~0.84 GB random HBM reads).
  pool_kernel<<<(BATCH * 2) / 8, 256, 0, stream>>>(ids, emb, X0);

  // MLP: grid.x = M / (16 rows * 8 waves) = 128; grid.y = N/64.
  gemm_bias_act<1><<<dim3(128, 8), 256, 0, stream>>>(X0, Wb1, b1, X1, BATCH, 512, 256);
  gemm_bias_act<1><<<dim3(128, 4), 256, 0, stream>>>(X1, Wb2, b2, X2, BATCH, 256, 512);
  gemm_bias_act<1><<<dim3(128, 2), 256, 0, stream>>>(X2, Wb3, b3, X3, BATCH, 128, 256);
  gemm_bias_act<1><<<dim3(128, 1), 256, 0, stream>>>(X3, Wb4, b4, X4, BATCH,  64, 128);

  // Final dot + sigmoid -> fp32 output.
  final_layer<<<(BATCH + 255) / 256, 256, 0, stream>>>(X4, W5, b5, out, BATCH);
}
